// WITRAN_2DPSGMU_Encoder_24034636988650
// MI455X (gfx1250) — matrix operand order repaired and checked by reading, not compiled, not run on hardware
//
#include <hip/hip_runtime.h>
#include <hip/hip_bf16.h>

// ---------------- problem constants ----------------
#define HID   256
#define NB    32       // batch B
#define RR    24       // cols R
#define LORIG 48       // rows
#define LL    71       // R + Lorig - 1
#define CIN   32       // input channels
#define MM    768      // R*B  (cells per step)
#define KK    544      // 2H + C
#define NN    1536     // 6H

// output layout (floats): output_all | hidden_col_all | hidden_row_all
#define OUT_ALL_SZ  ((size_t)MM * LL * 2 * HID)         // 27,918,336
#define HC_OFF      OUT_ALL_SZ
#define HR_OFF      (OUT_ALL_SZ + (size_t)NB * LORIG * HID)

// ---------------- WMMA types ----------------
typedef __attribute__((ext_vector_type(16))) __bf16        v16bf;
typedef __attribute__((ext_vector_type(8)))  float         v8f;
typedef __attribute__((ext_vector_type(4)))  unsigned int  v4u;

union FragBF { v4u q[2]; v16bf v; };

__device__ __forceinline__ unsigned short f32_to_bf16(float f) {
    unsigned int u = __float_as_uint(f);
    u += 0x7fffu + ((u >> 16) & 1u);          // round-to-nearest-even
    return (unsigned short)(u >> 16);
}

__device__ __forceinline__ float sigm(float x) {
    return 1.0f / (1.0f + __expf(-x));
}
__device__ __forceinline__ float tanh_fast(float x) {
    return 2.0f / (1.0f + __expf(-2.0f * x)) - 1.0f;
}

// ---------------- one-time: W fp32 -> bf16 ----------------
__global__ void __launch_bounds__(256)
pack_w_kernel(const float* __restrict__ W, unsigned short* __restrict__ Wb) {
    int t = blockIdx.x * blockDim.x + threadIdx.x;
    if (t < NN * KK) Wb[t] = f32_to_bf16(W[t]);
}

// ---------------- one-time: zero states, build Zb0 for s=0 ----------------
__global__ void __launch_bounds__(256)
init_kernel(const float* __restrict__ inp, unsigned short* __restrict__ Zb0,
            float* __restrict__ hrow, float* __restrict__ hc0, float* __restrict__ hc1) {
    int t = blockIdx.x * blockDim.x + threadIdx.x;
    if (t < MM * KK) {
        int m = t / KK, k = t % KK;
        unsigned short v = 0;
        if (k >= 2 * HID) {                 // x slice for s = 0: only r == 0 valid
            int c = k - 2 * HID;
            int r = m >> 5, b = m & 31;
            if (r == 0)
                v = f32_to_bf16(inp[(size_t)b * (LORIG * RR * CIN) + c]); // l0=0, r=0
        }
        Zb0[t] = v;
    }
    if (t < MM * HID) { hrow[t] = 0.f; hc0[t] = 0.f; hc1[t] = 0.f; }
}

// ---------------- fused per-step kernel ----------------
// One wave owns a 32(M) x 16(N) tile and accumulates ALL SIX gate groups for it.
// In the WMMA C-layout the six gate values of a given (row,col) land on the SAME
// lane / same accumulator element across group fragments, so the full PSGMU
// gating runs in-register with no LDS and no gate buffer.
//
// grid = (2 N-halves, 24 M-tiles), block = 256 threads (8 waves, 16 N-cols each).
__global__ void __launch_bounds__(256)
witran_step_kernel(const unsigned short* __restrict__ Zb_cur,
                   unsigned short* __restrict__ Zb_nxt,
                   const unsigned short* __restrict__ Wb,
                   const float* __restrict__ Bp,
                   const float* __restrict__ inp,
                   float* __restrict__ hrow,
                   const float* __restrict__ hcol_in,
                   float* __restrict__ hcol_out,
                   float* __restrict__ out, int s) {
    const int lane = threadIdx.x & 31;
    const int wave = threadIdx.x >> 5;
    const int lm   = lane & 15;
    const int lh   = lane >> 4;
    const int m0   = blockIdx.y * 32;                // block/wave M tile: 32 rows
    const int n0   = blockIdx.x * 128 + wave * 16;   // wave's 16 cols inside each group

    v8f acc[6][2];
    {
        v8f zero = {0.f, 0.f, 0.f, 0.f, 0.f, 0.f, 0.f, 0.f};
        #pragma unroll
        for (int g = 0; g < 6; ++g) { acc[g][0] = zero; acc[g][1] = zero; }
    }

    for (int kb = 0; kb < KK; kb += 32) {
        // A fragments (16x32 bf16): row = M (lane 0-15), K split {0-7,16-23}/{8-15,24-31}
        FragBF a[2];
        #pragma unroll
        for (int mf = 0; mf < 2; ++mf) {
            const unsigned short* p = Zb_cur + (size_t)(m0 + mf * 16 + lm) * KK + kb + lh * 8;
            a[mf].q[0] = *(const v4u*)p;            // K = kb + lh*8      .. +7
            a[mf].q[1] = *(const v4u*)(p + 16);     // K = kb + 16 + lh*8 .. +7
        }
        // B fragments (32x16 bf16), one per gate group; B[k][n] = Wb[n][k]
        FragBF b[6];
        #pragma unroll
        for (int g = 0; g < 6; ++g) {
            const unsigned short* p = Wb + (size_t)(g * HID + n0 + lm) * KK + kb + lh * 8;
            b[g].q[0] = *(const v4u*)p;             // K = kb + lh*8      .. +7
            b[g].q[1] = *(const v4u*)(p + 16);      // K = kb + 16 + lh*8 .. +7
        }
        if (kb + 32 < KK) {                         // global_prefetch_b8: next W K-block
            __builtin_prefetch(Wb + (size_t)(n0 + lm) * KK + kb + 32, 0, 3);
        }
        #pragma unroll
        for (int g = 0; g < 6; ++g)
            #pragma unroll
            for (int mf = 0; mf < 2; ++mf)
                acc[g][mf] = __builtin_amdgcn_wmma_f32_16x16x32_bf16(
                    false, a[mf].v, false, b[g].v, (short)0, acc[g][mf], false, false);
    }

    // ---- fused epilogue: all six gate values for (row,col) sit on this lane ----
    const int col = n0 + lm;
    float bp[6];
    #pragma unroll
    for (int g = 0; g < 6; ++g) bp[g] = Bp[g * HID + col];

    #pragma unroll
    for (int mf = 0; mf < 2; ++mf) {
        #pragma unroll
        for (int e = 0; e < 8; ++e) {
            const int row = m0 + mf * 16 + lh * 8 + e;     // C-layout: M = e + lh*8
            const int r = row >> 5, b = row & 31;
            const float msk = ((r <= s) && (s < RR)) ? 1.0f : 0.0f;

            const float u_row = sigm(acc[0][mf][e] + msk * bp[0]);
            const float o_row = sigm(acc[1][mf][e] + msk * bp[1]);
            const float u_col = sigm(acc[2][mf][e] + msk * bp[2]);
            const float o_col = sigm(acc[3][mf][e] + msk * bp[3]);
            const float i_row = tanh_fast(acc[4][mf][e] + msk * bp[4]);
            const float i_col = tanh_fast(acc[5][mf][e] + msk * bp[5]);

            const float hr = hrow[(size_t)row * HID + col];
            const float hc = hcol_in[(size_t)row * HID + col];
            const float hr_new = tanh_fast((1.f - u_row) * hr + u_row * i_row) * o_row;
            const float hc_new = tanh_fast((1.f - u_col) * hc + u_col * i_col) * o_col;

            int mr = row + NB; if (mr >= MM) mr -= MM;     // h_col roll by B
            hrow[(size_t)row * HID + col]    = hr_new;
            hcol_out[(size_t)mr * HID + col] = hc_new;

            const size_t ob = (size_t)row * (LL * 2 * HID) + (size_t)s * (2 * HID);
            out[ob + col]       = hr_new;
            out[ob + HID + col] = hc_new;

            if (s >= RR - 1 && r == RR - 1)
                out[HC_OFF + ((size_t)b * LORIG + (s - (RR - 1))) * HID + col] = hc_new;
            if (s >= LORIG - 1 && r == s - (LORIG - 1))
                out[HR_OFF + ((size_t)b * RR + (s - (LORIG - 1))) * HID + col] = hr_new;

            // repack bf16 Z for step s+1 (ping-pong buffer: no cross-block races)
            Zb_nxt[(size_t)row * KK + col]      = f32_to_bf16(hr_new);
            Zb_nxt[(size_t)mr  * KK + HID + col] = f32_to_bf16(hc_new);
        }
    }

    // x slice of Z for step s+1 (one writer per M tile: the blockIdx.x==0 half)
    if (blockIdx.x == 0 && s + 1 < LL) {
        for (int idx = threadIdx.x; idx < NB * CIN; idx += 256) {   // 32 rows x 32 ch
            const int ml = idx >> 5, c = idx & 31;
            const int row = m0 + ml;
            const int r = row >> 5, b = row & 31;
            const int l0 = (s + 1) - r;
            float xv = 0.f;
            if (l0 >= 0 && l0 < LORIG)
                xv = inp[(((size_t)b * LORIG + l0) * RR + r) * CIN + c];
            Zb_nxt[(size_t)row * KK + 2 * HID + c] = f32_to_bf16(xv);
        }
    }
}

// ---------------- host ----------------
extern "C" void kernel_launch(void* const* d_in, const int* in_sizes, int n_in,
                              void* d_out, int out_size, void* d_ws, size_t ws_size,
                              hipStream_t stream) {
    (void)in_sizes; (void)n_in; (void)out_size; (void)ws_size;
    const float* inp = (const float*)d_in[0];   // (32,48,24,32)
    const float* W   = (const float*)d_in[1];   // (1536,544)
    const float* Bp  = (const float*)d_in[2];   // (1536,)
    float* out = (float*)d_out;

    char* ws = (char*)d_ws;
    const size_t WB_B = (size_t)NN * KK * sizeof(unsigned short);   // 1,671,168
    const size_t ZB_B = (size_t)MM * KK * sizeof(unsigned short);   //   835,584
    const size_t ST_B = (size_t)MM * HID * sizeof(float);           //   786,432

    unsigned short* Wb  = (unsigned short*)(ws);
    unsigned short* Zb0 = (unsigned short*)(ws + WB_B);
    unsigned short* Zb1 = (unsigned short*)(ws + WB_B + ZB_B);
    float*          hrow = (float*)(ws + WB_B + 2 * ZB_B);
    float*          hc0  = (float*)(ws + WB_B + 2 * ZB_B + ST_B);
    float*          hc1  = (float*)(ws + WB_B + 2 * ZB_B + 2 * ST_B);

    pack_w_kernel<<<(NN * KK + 255) / 256, 256, 0, stream>>>(W, Wb);
    init_kernel<<<(MM * KK + 255) / 256, 256, 0, stream>>>(inp, Zb0, hrow, hc0, hc1);

    dim3 grid(2, MM / 32);   // 2 N-halves x 24 M-tiles
    for (int s = 0; s < LL; ++s) {
        unsigned short* zin  = (s & 1) ? Zb1 : Zb0;
        unsigned short* zout = (s & 1) ? Zb0 : Zb1;
        float* hin  = (s & 1) ? hc1 : hc0;
        float* hout = (s & 1) ? hc0 : hc1;
        witran_step_kernel<<<grid, 256, 0, stream>>>(zin, zout, Wb, Bp, inp,
                                                     hrow, hin, hout, out, s);
    }
}
